// MultiHeadedAttention_14396730376767
// MI455X (gfx1250) — matrix operand order repaired and checked by reading, not compiled, not run on hardware
//
#include <hip/hip_runtime.h>
#include <hip/hip_bf16.h>

// MI455X / gfx1250, wave32. bf16 WMMA (16x16x32) with f32 accumulation.
// Q/K GEMM -> bf16, V GEMM -> bf16 transposed per head,
// flash attention (async K staging into LDS, online softmax) -> bf16,
// output GEMM -> f32. Workspace: 32 MB.

typedef __bf16 bf16;
typedef __attribute__((ext_vector_type(16))) __bf16 v16bf;
typedef __attribute__((ext_vector_type(8)))  __bf16 v8bf;
typedef __attribute__((ext_vector_type(8)))  float  v8f;

#define HH 16
#define DM 1024
#define DKH 64
#define BBATCH 2
#define SSEQ 2048
#define MTOT (BBATCH*SSEQ)   // 4096

// ---- raw loads (no conversion) + deferred conversion to bf16 frags ---------
template <typename T> struct RV;
template <> struct RV<float> { typedef v8f  t; };
template <> struct RV<bf16>  { typedef v8bf t; };

__device__ __forceinline__ void ldraw(const float* p, int off2, v8f& x, v8f& y) {
    x = *(const v8f*)p; y = *(const v8f*)(p + off2);
}
__device__ __forceinline__ void ldraw(const bf16* p, int off2, v8bf& x, v8bf& y) {
    x = *(const v8bf*)p; y = *(const v8bf*)(p + off2);
}
__device__ __forceinline__ v16bf cvt16(v8f x, v8f y) {
    v16bf r;
#pragma unroll
    for (int t = 0; t < 8; ++t) { r[t] = (bf16)x[t]; r[t + 8] = (bf16)y[t]; }
    return r;
}
__device__ __forceinline__ v16bf cvt16(v8bf x, v8bf y) {
    v16bf r;
#pragma unroll
    for (int t = 0; t < 8; ++t) { r[t] = x[t]; r[t + 8] = y[t]; }
    return r;
}
__device__ __forceinline__ v16bf mk16(v8bf lo, v8bf hi) { return cvt16(lo, hi); }

// ---- 16-lane (half-wave) row reductions, wave32 ----------------------------
__device__ __forceinline__ v8f red_max16(v8f t) {
#pragma unroll
    for (int m = 1; m < 16; m <<= 1) {
#pragma unroll
        for (int r = 0; r < 8; ++r) t[r] = fmaxf(t[r], __shfl_xor(t[r], m, 32));
    }
    return t;
}
__device__ __forceinline__ v8f red_sum16(v8f t) {
#pragma unroll
    for (int m = 1; m < 16; m <<= 1) {
#pragma unroll
        for (int r = 0; r < 8; ++r) t[r] += __shfl_xor(t[r], m, 32);
    }
    return t;
}

// ---- CDNA5 async global->LDS staging (ASYNCcnt) ----------------------------
// Copies a 64x64 bf16 tile (8KB): 512 x 16B chunks, 16 per lane.
__device__ __forceinline__ void stage_tile_async(const bf16* __restrict__ gbase,
                                                 size_t row_stride, bf16* lbase, int lane) {
#pragma unroll
    for (int c = 0; c < 16; ++c) {
        const int id  = c * 32 + lane;
        const int row = id >> 3;            // 8 x 16B chunks per 128B row
        const int col = (id & 7) * 8;       // elements
        const bf16* g = gbase + (size_t)row * row_stride + col;
        const unsigned l = (unsigned)(size_t)(lbase + row * DKH + col);
        asm volatile("global_load_async_to_lds_b128 %0, %1, off"
                     :: "v"(l), "v"(g) : "memory");
    }
}
__device__ __forceinline__ void wait_async_le16() {
    asm volatile("s_wait_asynccnt 0x10" ::: "memory");
}

// ---- GEMM: out[m,n] = sum_k A[m,k] * W[n,k] + bias[n]  (X @ W^T + b) -------
// M=4096, N=1024, K=1024. One wave computes a 64x64 tile (4x4 WMMA accums).
// Software-pipelined: raw loads (k+32) -> 16 WMMAs (k) -> cvt (k+32).
// OUT_MODE: 0 = f32 row-major, 1 = bf16 row-major,
//           2 = bf16 transposed per head: Vt[((b*H+h)*DKH+d)*S + s]
template <typename TA, int OUT_MODE>
__global__ __launch_bounds__(128, 1) void wmma_gemm_xwT(
    const TA* __restrict__ A, const float* __restrict__ W,
    const float* __restrict__ bias, void* __restrict__ out)
{
    const int lane = threadIdx.x & 31;
    const int half = lane >> 4, lr = lane & 15;
    const int w  = blockIdx.x * 4 + (threadIdx.x >> 5);
    const int m0 = (w >> 4) * 64;          // 64 m-tiles
    const int n0 = (w & 15) * 64;          // 16 n-tiles

    typedef typename RV<TA>::t rawA;
    const TA*    pa0[4];
    const float* pb0[4];
#pragma unroll
    for (int i = 0; i < 4; ++i) pa0[i] = A + (size_t)(m0 + 16 * i + lr) * DM + half * 8;
#pragma unroll
    for (int j = 0; j < 4; ++j) pb0[j] = W + (size_t)(n0 + 16 * j + lr) * DM + half * 8;

    v8f c[4][4] = {};

    // prologue: k0 = 0 fragments
    v16bf afr[4], bfr[4];
    {
        rawA ax, ay; v8f bx, by;
#pragma unroll
        for (int i = 0; i < 4; ++i) { ldraw(pa0[i], 16, ax, ay); afr[i] = cvt16(ax, ay); }
#pragma unroll
        for (int j = 0; j < 4; ++j) { ldraw(pb0[j], 16, bx, by); bfr[j] = cvt16(bx, by); }
    }

    for (int k0 = 0; k0 < DM; k0 += 32) {
        const int kn = (k0 + 32 < DM) ? (k0 + 32) : k0;   // last iter reloads (discarded)

        // 1) issue next-slice raw loads (no conversion yet)
        rawA axn[4], ayn[4]; v8f bxn[4], byn[4];
#pragma unroll
        for (int j = 0; j < 4; ++j) ldraw(pb0[j] + kn, 16, bxn[j], byn[j]);
#pragma unroll
        for (int i = 0; i < 4; ++i) ldraw(pa0[i] + kn, 16, axn[i], ayn[i]);
        if (k0 + 96 < DM)
            __builtin_prefetch(W + (size_t)(n0 + lr) * DM + k0 + 96, 0, 3);

        // 2) 16 WMMAs on current fragments (independent of the new loads)
#pragma unroll
        for (int j = 0; j < 4; ++j)
#pragma unroll
            for (int i = 0; i < 4; ++i)
                c[i][j] = __builtin_amdgcn_wmma_f32_16x16x32_bf16(
                    false, afr[i], false, bfr[j], (short)0, c[i][j], false, false);

        // 3) convert next-slice raw data into fragments
#pragma unroll
        for (int i = 0; i < 4; ++i) afr[i] = cvt16(axn[i], ayn[i]);
#pragma unroll
        for (int j = 0; j < 4; ++j) bfr[j] = cvt16(bxn[j], byn[j]);
    }

#pragma unroll
    for (int j = 0; j < 4; ++j) {
        const int n = n0 + 16 * j + lr;
        const float bj = bias[n];
#pragma unroll
        for (int i = 0; i < 4; ++i) {
#pragma unroll
            for (int r = 0; r < 8; ++r) {
                const int m = m0 + 16 * i + r + 8 * half;   // C layout
                const float v = c[i][j][r] + bj;
                if (OUT_MODE == 0) {
                    ((float*)out)[(size_t)m * DM + n] = v;
                } else if (OUT_MODE == 1) {
                    ((bf16*)out)[(size_t)m * DM + n] = (bf16)v;
                } else {
                    const int h = n >> 6, d = n & 63;
                    const int b = m >> 11, kp = m & (SSEQ - 1);
                    ((bf16*)out)[((size_t)((b * HH + h) * DKH + d)) * SSEQ + kp] = (bf16)v;
                }
            }
        }
    }
}

// ---- Flash attention: one wave owns 32 q-rows of one (b,h) -----------------
// 2 waves / block. K tiles double-buffered in LDS via async DMA loads.
__global__ __launch_bounds__(64, 1) void flash_attn(
    const bf16* __restrict__ Qb, const bf16* __restrict__ Kb,
    const bf16* __restrict__ Vt, const int* __restrict__ mask,
    bf16* __restrict__ Ob)
{
    __shared__ __align__(32) bf16 kstage[2][2][64 * 64];  // [wave][buf] 8KB tiles
    __shared__ __align__(32) bf16 pbuf[2][32 * 64];       // per-wave P relayout tile
    const int lane = threadIdx.x & 31, half = lane >> 4, lr = lane & 15;
    const int widx = threadIdx.x >> 5;
    const int w  = blockIdx.x * 2 + widx;
    const int qt = w & 63;               // S/32 = 64 q tiles
    const int h  = (w >> 6) & 15;
    const int b  = w >> 10;
    const int q0 = qt * 32;
    bf16* pl = pbuf[widx];

    // Q fragments: rows q0..q0+31, d_k split into two K=32 steps
    v16bf qf[2][2];
#pragma unroll
    for (int i = 0; i < 2; ++i)
#pragma unroll
        for (int ks = 0; ks < 2; ++ks) {
            const bf16* p = Qb + (size_t)(b * SSEQ + q0 + 16 * i + lr) * DM
                          + h * DKH + 32 * ks + half * 16;
            qf[i][ks] = mk16(*(const v8bf*)p, *(const v8bf*)(p + 8));
        }

    v8f oacc[2][4] = {};
    v8f mrun[2], lrun[2];
#pragma unroll
    for (int i = 0; i < 2; ++i)
#pragma unroll
        for (int r = 0; r < 8; ++r) { mrun[i][r] = -3.0e38f; lrun[i][r] = 0.0f; }

    const bf16* kgbase = Kb + (size_t)(b * SSEQ) * DM + h * DKH;
    const bf16* vgbase = Vt + (size_t)((b * HH + h) * DKH) * SSEQ;

    // stage k-tile 0 into buffer 0
    stage_tile_async(kgbase, DM, &kstage[widx][0][0], lane);
    int buf = 0;

    for (int kt = 0; kt < SSEQ / 64; ++kt) {
        const int kb = kt * 64;
        const int kbn = (kt + 1 < SSEQ / 64) ? kb + 64 : kb;

        // issue async staging of the next K tile, then wait for current one
        stage_tile_async(kgbase + (size_t)kbn * DM, DM, &kstage[widx][buf ^ 1][0], lane);
        wait_async_le16();   // in-order completion: current tile resident
        const bf16* kst = &kstage[widx][buf][0];

        // scores: S = Q @ K^T (K fragments from LDS)
        v16bf kf[2][4];
#pragma unroll
        for (int ks = 0; ks < 2; ++ks)
#pragma unroll
            for (int j = 0; j < 4; ++j)
                kf[ks][j] = *(const v16bf*)(kst + (16 * j + lr) * DKH + 32 * ks + 16 * half);
        v8f s[2][4] = {};
#pragma unroll
        for (int ks = 0; ks < 2; ++ks)
#pragma unroll
            for (int j = 0; j < 4; ++j)
#pragma unroll
                for (int i = 0; i < 2; ++i)
                    s[i][j] = __builtin_amdgcn_wmma_f32_16x16x32_bf16(
                        false, qf[i][ks], false, kf[ks][j], (short)0, s[i][j], false, false);

        // issue V fragment loads now; consumed after softmax (latency hidden)
        v16bf vf[2][4];
#pragma unroll
        for (int kk = 0; kk < 2; ++kk)
#pragma unroll
            for (int jd = 0; jd < 4; ++jd) {
                const bf16* pv = vgbase + (size_t)(16 * jd + lr) * SSEQ
                               + kb + 32 * kk + half * 16;
                vf[kk][jd] = *(const v16bf*)pv;
            }

        // scale 1/sqrt(64) + mask bias (per key column)
        float mb[4];
#pragma unroll
        for (int j = 0; j < 4; ++j)
            mb[j] = (mask[b * SSEQ + kb + 16 * j + lr] == 0) ? -1.0e9f : 0.0f;
#pragma unroll
        for (int i = 0; i < 2; ++i)
#pragma unroll
            for (int j = 0; j < 4; ++j)
#pragma unroll
                for (int r = 0; r < 8; ++r)
                    s[i][j][r] = s[i][j][r] * 0.125f + mb[j];

        // online softmax per 16-row group
#pragma unroll
        for (int i = 0; i < 2; ++i) {
            v8f t = s[i][0];
#pragma unroll
            for (int r = 0; r < 8; ++r)
                t[r] = fmaxf(fmaxf(t[r], s[i][1][r]), fmaxf(s[i][2][r], s[i][3][r]));
            t = red_max16(t);
            v8f mnew, scale;
#pragma unroll
            for (int r = 0; r < 8; ++r) {
                mnew[r]  = fmaxf(mrun[i][r], t[r]);
                scale[r] = __expf(mrun[i][r] - mnew[r]);
            }
            v8f rs;
#pragma unroll
            for (int r = 0; r < 8; ++r) rs[r] = 0.0f;
#pragma unroll
            for (int j = 0; j < 4; ++j)
#pragma unroll
                for (int r = 0; r < 8; ++r) {
                    const float pv = __expf(s[i][j][r] - mnew[r]);
                    s[i][j][r] = pv;
                    rs[r] += pv;
                }
            rs = red_sum16(rs);
#pragma unroll
            for (int r = 0; r < 8; ++r) {
                lrun[i][r] = lrun[i][r] * scale[r] + rs[r];
                mrun[i][r] = mnew[r];
            }
#pragma unroll
            for (int jd = 0; jd < 4; ++jd)
#pragma unroll
                for (int r = 0; r < 8; ++r)
                    oacc[i][jd][r] *= scale[r];

            // P (C-layout) -> LDS, bf16
#pragma unroll
            for (int j = 0; j < 4; ++j)
#pragma unroll
                for (int r = 0; r < 8; ++r)
                    pl[(16 * i + r + 8 * half) * 64 + 16 * j + lr] = (bf16)s[i][j][r];
        }

        // reload P as A-fragments (wave-private LDS; DS ordered within wave)
        v16bf pf[2][2];
#pragma unroll
        for (int i = 0; i < 2; ++i)
#pragma unroll
            for (int kk = 0; kk < 2; ++kk) {
                const bf16* p = pl + (16 * i + lr) * 64 + 32 * kk + half * 16;
                pf[i][kk] = mk16(*(const v8bf*)p, *(const v8bf*)(p + 8));
            }

        // O += P @ V
#pragma unroll
        for (int kk = 0; kk < 2; ++kk)
#pragma unroll
            for (int jd = 0; jd < 4; ++jd)
#pragma unroll
                for (int i = 0; i < 2; ++i)
                    oacc[i][jd] = __builtin_amdgcn_wmma_f32_16x16x32_bf16(
                        false, pf[i][kk], false, vf[kk][jd], (short)0, oacc[i][jd], false, false);

        buf ^= 1;
    }

    // normalize and store O (bf16, row-major (B*S, D))
#pragma unroll
    for (int i = 0; i < 2; ++i) {
        v8f inv;
#pragma unroll
        for (int r = 0; r < 8; ++r) inv[r] = 1.0f / lrun[i][r];
#pragma unroll
        for (int jd = 0; jd < 4; ++jd)
#pragma unroll
            for (int r = 0; r < 8; ++r) {
                const int row = b * SSEQ + q0 + 16 * i + r + 8 * half;
                const int col = h * DKH + 16 * jd + lr;
                Ob[(size_t)row * DM + col] = (bf16)(oacc[i][jd][r] * inv[r]);
            }
    }
}

// ---- launch ----------------------------------------------------------------
extern "C" void kernel_launch(void* const* d_in, const int* in_sizes, int n_in,
                              void* d_out, int out_size, void* d_ws, size_t ws_size,
                              hipStream_t stream) {
    const float* query = (const float*)d_in[0];
    const float* key   = (const float*)d_in[1];
    const float* value = (const float*)d_in[2];
    const int*   mask  = (const int*)d_in[3];
    const float* wq = (const float*)d_in[4];  const float* bq = (const float*)d_in[5];
    const float* wk = (const float*)d_in[6];  const float* bk = (const float*)d_in[7];
    const float* wv = (const float*)d_in[8];  const float* bv = (const float*)d_in[9];
    const float* wo = (const float*)d_in[10]; const float* bo = (const float*)d_in[11];

    char* ws = (char*)d_ws;                       // 32 MB used
    const size_t seg = (size_t)MTOT * DM * sizeof(bf16);   // 8 MB
    bf16* Qb = (bf16*)(ws);
    bf16* Kb = (bf16*)(ws + seg);
    bf16* Vt = (bf16*)(ws + 2 * seg);
    bf16* Ob = (bf16*)(ws + 3 * seg);

    const int gemm_blocks = (MTOT / 64) * (DM / 64) / 4;   // 256 blocks, 1024 waves
    wmma_gemm_xwT<float, 1><<<gemm_blocks, dim3(128), 0, stream>>>(query, wq, bq, Qb);
    wmma_gemm_xwT<float, 1><<<gemm_blocks, dim3(128), 0, stream>>>(key,   wk, bk, Kb);
    wmma_gemm_xwT<float, 2><<<gemm_blocks, dim3(128), 0, stream>>>(value, wv, bv, Vt);

    const int attn_blocks = (BBATCH * HH * (SSEQ / 32)) / 2;  // 1024 blocks, 2 waves each
    flash_attn<<<attn_blocks, dim3(64), 0, stream>>>(Qb, Kb, Vt, mask, Ob);

    wmma_gemm_xwT<bf16, 0><<<gemm_blocks, dim3(128), 0, stream>>>(Ob, wo, bo, (float*)d_out);
}
